// MotionPredModule_47983374631456
// MI455X (gfx1250) — matrix operand order repaired and checked by reading, not compiled, not run on hardware
//
#include <hip/hip_runtime.h>
#include <hip/hip_bf16.h>
#include <math.h>

#define NPTS 4096
#define DIMF 128
#define HIDF 256

typedef __attribute__((ext_vector_type(16))) _Float16 v16h;
typedef __attribute__((ext_vector_type(8)))  _Float16 v8h;
typedef __attribute__((ext_vector_type(8)))  float    v8f;
typedef _Float16 half_t;

// ---------- small helpers ----------
__device__ inline unsigned fmap(float f) {
    unsigned u = __float_as_uint(f);
    return (u >> 31) ? ~u : (u | 0x80000000u);
}
__device__ inline float funmap(unsigned m) {
    return __uint_as_float((m >> 31) ? (m ^ 0x80000000u) : ~m);
}

__global__ void k_fill(float* p, float v, int n) {
    int i = blockIdx.x * blockDim.x + threadIdx.x;
    if (i < n) p[i] = v;
}
__global__ void k_cvt_h(const float* __restrict__ s, half_t* __restrict__ d, int n) {
    int i = blockIdx.x * blockDim.x + threadIdx.x;
    if (i < n) d[i] = (half_t)s[i];
}
// f32 W[K][N] -> f16 WT[N][K]
__global__ void k_cvt_t(const float* __restrict__ W, half_t* __restrict__ WT, int K, int N) {
    int id = blockIdx.x * blockDim.x + threadIdx.x;
    if (id >= K * N) return;
    int k = id / N, n = id % N;
    WT[(size_t)n * K + k] = (half_t)W[id];
}

// ---------- stage 0: data_x = [x1 | (x2^T @ Wk + bk)] ----------
__global__ void k_build(const float* __restrict__ x1, const float* __restrict__ x2,
                        const float* __restrict__ Wk, const float* __restrict__ bk,
                        float* __restrict__ dx8) {
    int n = blockIdx.x * blockDim.x + threadIdx.x;
    if (n >= NPTS) return;
    for (int c = 0; c < 4; ++c) {
        float s = bk[0];
        for (int k = 0; k < 8; ++k) s += x2[(n * 8 + k) * 4 + c] * Wk[k];
        dx8[n * 8 + c]     = x1[n * 4 + c];
        dx8[n * 8 + 4 + c] = s;
    }
}

// ---------- stage 1: pairwise distances + row max ----------
__global__ void k_dist(const float* __restrict__ dx8, float* __restrict__ dist,
                       float* __restrict__ maxd) {
    int i = blockIdx.x, t = threadIdx.x;
    __shared__ float red[256];
    float xi = dx8[i * 8 + 0], yi = dx8[i * 8 + 1];
    float mx = 0.f;
    for (int j = t; j < NPTS; j += 256) {
        float dx = xi - dx8[j * 8 + 0];
        float dy = yi - dx8[j * 8 + 1];
        float d = sqrtf(dx * dx + dy * dy);
        dist[(size_t)i * NPTS + j] = d;
        mx = fmaxf(mx, d);
    }
    red[t] = mx; __syncthreads();
    for (int s = 128; s > 0; s >>= 1) {
        if (t < s) red[t] = fmaxf(red[t], red[t + s]);
        __syncthreads();
    }
    if (t == 0) maxd[i] = red[0];
}

// ---------- stage 2: 9-NN per row (ascending distance; prefixes give k=3,5) ----------
__global__ void k_knn(const float* __restrict__ dist, int* __restrict__ knn) {
    int i = blockIdx.x, t = threadIdx.x;
    __shared__ float sv[256];
    __shared__ int   si[256];
    __shared__ int   sel[9];
    const float* row = dist + (size_t)i * NPTS;
    for (int r = 0; r < 9; ++r) {
        float bv = 3.0e38f; int bi = NPTS;
        for (int j = t; j < NPTS; j += 256) {
            if (j == i) continue;
            bool used = false;
            for (int q = 0; q < r; ++q) if (sel[q] == j) used = true;
            if (used) continue;
            float v = row[j];
            if (v < bv || (v == bv && j < bi)) { bv = v; bi = j; }
        }
        sv[t] = bv; si[t] = bi; __syncthreads();
        for (int s = 128; s > 0; s >>= 1) {
            if (t < s) {
                if (sv[t + s] < sv[t] || (sv[t + s] == sv[t] && si[t + s] < si[t])) {
                    sv[t] = sv[t + s]; si[t] = si[t + s];
                }
            }
            __syncthreads();
        }
        if (t == 0) { sel[r] = si[0]; knn[i * 9 + r] = si[0]; }
        __syncthreads();
    }
}

// ---------- GCN pieces ----------
__global__ void k_deg(const int* __restrict__ knn, int k, float* degf) {
    int id = blockIdx.x * blockDim.x + threadIdx.x;
    if (id >= NPTS * k) return;
    int i = id / k, j = id % k;
    atomicAdd(&degf[knn[i * 9 + j]], 1.0f);
}
__global__ void k_dinv(const float* __restrict__ degf, float* __restrict__ dinv) {
    int i = blockIdx.x * blockDim.x + threadIdx.x;
    if (i < NPTS) dinv[i] = rsqrtf(degf[i]);
}
__global__ void k_gemm_f32(const float* __restrict__ X, const float* __restrict__ W,
                           float* __restrict__ Y, int K, int F) {
    int id = blockIdx.x * blockDim.x + threadIdx.x;
    if (id >= NPTS * F) return;
    int n = id / F, f = id % F;
    float s = 0.f;
    for (int k = 0; k < K; ++k) s += X[n * K + k] * W[k * F + f];
    Y[id] = s;
}
__global__ void k_agg(const float* __restrict__ xw, const int* __restrict__ knn, int k,
                      const float* __restrict__ dinv, float* out, int F) {
    int id = blockIdx.x * blockDim.x + threadIdx.x;
    if (id >= NPTS * (k + 1) * F) return;
    int f = id % F, e = id / F;
    int i = e / (k + 1), j = e - i * (k + 1);
    int dst = (j == k) ? i : knn[i * 9 + j];
    atomicAdd(&out[dst * F + f], dinv[i] * dinv[dst] * xw[i * F + f]);
}
__global__ void k_bias_relu(const float* __restrict__ x, const float* __restrict__ b,
                            float* __restrict__ y, int F) {
    int id = blockIdx.x * blockDim.x + threadIdx.x;
    if (id >= NPTS * F) return;
    y[id] = fmaxf(x[id] + b[id % F], 0.f);
}
__global__ void k_lsm(const float* __restrict__ agg, const float* __restrict__ b,
                      float* __restrict__ emb, int off) {
    int n = blockIdx.x, f = threadIdx.x;  // 128 threads
    __shared__ float red[128];
    float o = agg[n * 128 + f] + b[f];
    red[f] = o; __syncthreads();
    for (int s = 64; s > 0; s >>= 1) {
        if (f < s) red[f] = fmaxf(red[f], red[f + s]);
        __syncthreads();
    }
    float m = red[0]; __syncthreads();
    red[f] = expf(o - m); __syncthreads();
    for (int s = 64; s > 0; s >>= 1) {
        if (f < s) red[f] += red[f + s];
        __syncthreads();
    }
    emb[n * 384 + off + f] = o - m - logf(red[0]);
}

// ---------- WMMA f16 GEMM: C = act(scale*(A @ Bt^T) + bias) ----------
// A:  M x K row-major f16
// Bt: N x K row-major f16 (i.e. the operand already transposed -> contiguous K loads)
// Each wave computes NT consecutive 16x16 N-tiles, reusing one A fragment per k-step.
template <int NT, int RELU, int OUTF16, int HASBIAS>
__global__ void k_wmma(const half_t* __restrict__ A, const half_t* __restrict__ Bt,
                       const float* __restrict__ bias, float scale,
                       float* __restrict__ Cf, half_t* __restrict__ Ch,
                       int M, int Nn, int Kk) {
    int wave = (blockIdx.x * blockDim.x + threadIdx.x) >> 5;
    int lane = threadIdx.x & 31;
    int tilesN = (Nn >> 4) / NT;
    int tm = wave / tilesN, tn = wave - tm * tilesN;
    if (tm >= (M >> 4)) return;  // wave-uniform exit
    int m0 = tm << 4;
    int n0 = tn * (16 * NT);
    int lr = lane & 15;
    bool hi = lane >= 16;
    int kbA = hi ? 8 : 0;    // A: lanes16-31 hold K 8..15 / 24..31
    int kbB = hi ? 8 : 0;    // B: lanes16-31 hold K 8..15 / 24..31
    v8f acc[NT] = {};
    const half_t* Arow = A + (size_t)(m0 + lr) * Kk;
    const half_t* Brow[NT];
#pragma unroll
    for (int t = 0; t < NT; ++t) Brow[t] = Bt + (size_t)(n0 + t * 16 + lr) * Kk;
    for (int kk = 0; kk < Kk; kk += 32) {
        __builtin_prefetch(Arow + kk + 64, 0, 3);
        v8h alo = *(const v8h*)(Arow + kk + kbA);
        v8h ahi = *(const v8h*)(Arow + kk + 16 + kbA);
        v16h a = __builtin_shufflevector(alo, ahi, 0, 1, 2, 3, 4, 5, 6, 7, 8, 9, 10, 11, 12,
                                         13, 14, 15);
#pragma unroll
        for (int t = 0; t < NT; ++t) {
            v8h blo = *(const v8h*)(Brow[t] + kk + kbB);
            v8h bhi = *(const v8h*)(Brow[t] + kk + 16 + kbB);
            v16h b = __builtin_shufflevector(blo, bhi, 0, 1, 2, 3, 4, 5, 6, 7, 8, 9, 10, 11,
                                             12, 13, 14, 15);
            acc[t] = __builtin_amdgcn_wmma_f32_16x16x32_f16(false, a, false, b, (short)0,
                                                            acc[t], false, false);
        }
    }
#pragma unroll
    for (int t = 0; t < NT; ++t) {
        int n = n0 + t * 16 + lr;
        float bv = HASBIAS ? bias[n] : 0.f;
#pragma unroll
        for (int v = 0; v < 8; ++v) {
            int m = m0 + v + (hi ? 8 : 0);
            float val = acc[t][v] * scale + bv;
            if (RELU) val = fmaxf(val, 0.f);
            if (OUTF16) Ch[(size_t)m * Nn + n] = (half_t)val;
            else        Cf[(size_t)m * Nn + n] = val;
        }
    }
}

// ---------- A = scores / rowmax ----------
__global__ void k_rowdiv(float* __restrict__ Ab) {
    __shared__ float row[NPTS];
    __shared__ float red[256];
    int i = blockIdx.x, t = threadIdx.x;
    float mx = -3.0e38f;
    for (int j = t; j < NPTS; j += 256) {
        float v = Ab[(size_t)i * NPTS + j];
        row[j] = v; mx = fmaxf(mx, v);
    }
    red[t] = mx; __syncthreads();
    for (int s = 128; s > 0; s >>= 1) {
        if (t < s) red[t] = fmaxf(red[t], red[t + s]);
        __syncthreads();
    }
    float m = red[0];
    for (int j = t; j < NPTS; j += 256) Ab[(size_t)i * NPTS + j] = row[j] / m;
}

// ---------- global min/max of A_cat (A, oushi, c3, c5, c9 computed on the fly) ----------
__global__ void k_mm_init(unsigned* mm) {
    mm[0] = fmap(3.0e38f);
    mm[1] = fmap(-3.0e38f);
}
__global__ void k_minmax(const float* __restrict__ Ab, const float* __restrict__ dist,
                         const float* __restrict__ maxd,
                         const float* wy3, const float* by3, const float* wx3, const float* bx3,
                         const float* wy5, const float* by5, const float* wx5, const float* bx5,
                         const float* wy9, const float* by9, const float* wx9, const float* bx9,
                         unsigned* mm) {
    int i = blockIdx.x, t = threadIdx.x;
    float y3[3], x3[3], y5[5], x5[5], y9[9], x9[9];
    for (int q = 0; q < 3; ++q) { y3[q] = wy3[q]; x3[q] = wx3[q]; }
    for (int q = 0; q < 5; ++q) { y5[q] = wy5[q]; x5[q] = wx5[q]; }
    for (int q = 0; q < 9; ++q) { y9[q] = wy9[q]; x9[q] = wx9[q]; }
    float b3 = by3[0] + bx3[0], b5 = by5[0] + bx5[0], b9 = by9[0] + bx9[0];
    float md = maxd[i];
    float mn = 3.0e38f, mx = -3.0e38f;
    for (int j = t; j < NPTS; j += 256) {
        float a = Ab[(size_t)i * NPTS + j];
        mn = fminf(mn, a); mx = fmaxf(mx, a);
        float ou = 1.f - dist[(size_t)i * NPTS + j] / md;
        mn = fminf(mn, ou); mx = fmaxf(mx, ou);
        float c;
        // k=3 (pad 1)
        c = b3;
        for (int q = 0; q < 3; ++q) {
            int jj = j + q - 1, ii = i + q - 1;
            if (jj >= 0 && jj < NPTS) c += y3[q] * Ab[(size_t)i * NPTS + jj];
            if (ii >= 0 && ii < NPTS) c += x3[q] * Ab[(size_t)ii * NPTS + j];
        }
        mn = fminf(mn, c); mx = fmaxf(mx, c);
        // k=5 (pad 2)
        c = b5;
        for (int q = 0; q < 5; ++q) {
            int jj = j + q - 2, ii = i + q - 2;
            if (jj >= 0 && jj < NPTS) c += y5[q] * Ab[(size_t)i * NPTS + jj];
            if (ii >= 0 && ii < NPTS) c += x5[q] * Ab[(size_t)ii * NPTS + j];
        }
        mn = fminf(mn, c); mx = fmaxf(mx, c);
        // k=9 (pad 4)
        c = b9;
        for (int q = 0; q < 9; ++q) {
            int jj = j + q - 4, ii = i + q - 4;
            if (jj >= 0 && jj < NPTS) c += y9[q] * Ab[(size_t)i * NPTS + jj];
            if (ii >= 0 && ii < NPTS) c += x9[q] * Ab[(size_t)ii * NPTS + j];
        }
        mn = fminf(mn, c); mx = fmaxf(mx, c);
    }
    __shared__ float smn[256], smx[256];
    smn[t] = mn; smx[t] = mx; __syncthreads();
    for (int s = 128; s > 0; s >>= 1) {
        if (t < s) {
            smn[t] = fminf(smn[t], smn[t + s]);
            smx[t] = fmaxf(smx[t], smx[t + s]);
        }
        __syncthreads();
    }
    if (t == 0) {
        atomicMin(&mm[0], fmap(smn[0]));
        atomicMax(&mm[1], fmap(smx[0]));
    }
}

// ---------- S1 = sum_j Ot[j,:] ----------
__global__ void k_s1(const float* __restrict__ dx8, float* __restrict__ S1) {
    __shared__ float red[4][256];
    int t = threadIdx.x;
    float a[4] = {0.f, 0.f, 0.f, 0.f};
    for (int n = t; n < NPTS; n += 256)
        for (int c = 0; c < 4; ++c) a[c] += dx8[n * 8 + 4 + c];
    for (int c = 0; c < 4; ++c) red[c][t] = a[c];
    __syncthreads();
    for (int s = 128; s > 0; s >>= 1) {
        if (t < s) for (int c = 0; c < 4; ++c) red[c][t] += red[c][t + s];
        __syncthreads();
    }
    if (t == 0) for (int c = 0; c < 4; ++c) S1[c] = red[c][0];
}

// ---------- OtC[j] = Ot[j] + sum_k corr(Ot, wy_k)[j] ----------
__global__ void k_otc(const float* __restrict__ dx8, const float* wy3, const float* wy5,
                      const float* wy9, float* __restrict__ OtC) {
    int j = blockIdx.x * blockDim.x + threadIdx.x;
    if (j >= NPTS) return;
    for (int c = 0; c < 4; ++c) {
        float s = dx8[j * 8 + 4 + c];
        for (int q = 0; q < 3; ++q) { int r = j + 1 - q; if (r >= 0 && r < NPTS) s += wy3[q] * dx8[r * 8 + 4 + c]; }
        for (int q = 0; q < 5; ++q) { int r = j + 2 - q; if (r >= 0 && r < NPTS) s += wy5[q] * dx8[r * 8 + 4 + c]; }
        for (int q = 0; q < 9; ++q) { int r = j + 4 - q; if (r >= 0 && r < NPTS) s += wy9[q] * dx8[r * 8 + 4 + c]; }
        OtC[j * 4 + c] = s;
    }
}

// ---------- per-row matvecs: Gg = A@Ot, GgC = A@OtC, Dg = dist@Ot ----------
__global__ void k_rowmm(const float* __restrict__ Ab, const float* __restrict__ dist,
                        const float* __restrict__ dx8, const float* __restrict__ OtC,
                        float* Gg, float* GgC, float* Dg) {
    int i = blockIdx.x, t = threadIdx.x;
    float acc[12];
    for (int v = 0; v < 12; ++v) acc[v] = 0.f;
    for (int j = t; j < NPTS; j += 256) {
        float a = Ab[(size_t)i * NPTS + j];
        float d = dist[(size_t)i * NPTS + j];
        for (int c = 0; c < 4; ++c) {
            float o = dx8[j * 8 + 4 + c];
            acc[c]     += a * o;
            acc[4 + c] += a * OtC[j * 4 + c];
            acc[8 + c] += d * o;
        }
    }
    __shared__ float red[12][256];
    for (int v = 0; v < 12; ++v) red[v][t] = acc[v];
    __syncthreads();
    for (int s = 128; s > 0; s >>= 1) {
        if (t < s) for (int v = 0; v < 12; ++v) red[v][t] += red[v][t + s];
        __syncthreads();
    }
    if (t == 0)
        for (int c = 0; c < 4; ++c) {
            Gg[i * 4 + c]  = red[c][0];
            GgC[i * 4 + c] = red[4 + c][0];
            Dg[i * 4 + c]  = red[8 + c][0];
        }
}

// ---------- raw -> normalized h (N x 4) ----------
__global__ void k_raw(const float* __restrict__ Gg, const float* __restrict__ GgC,
                      const float* __restrict__ Dg, const float* __restrict__ maxd,
                      const float* __restrict__ S1, const unsigned* __restrict__ mm,
                      const float* wx3, const float* wx5, const float* wx9,
                      const float* by3, const float* bx3, const float* by5, const float* bx5,
                      const float* by9, const float* bx9, float* __restrict__ h4) {
    int n = blockIdx.x * blockDim.x + threadIdx.x;
    if (n >= NPTS) return;
    float bsum = by3[0] + bx3[0] + by5[0] + bx5[0] + by9[0] + bx9[0];
    float mn = funmap(mm[0]), mx = funmap(mm[1]);
    float inv = 1.f / (mx - mn);
    float md = maxd[n];
    for (int c = 0; c < 4; ++c) {
        float cv = 0.f;
        for (int q = 0; q < 3; ++q) { int r = n + q - 1; if (r >= 0 && r < NPTS) cv += wx3[q] * Gg[r * 4 + c]; }
        for (int q = 0; q < 5; ++q) { int r = n + q - 2; if (r >= 0 && r < NPTS) cv += wx5[q] * Gg[r * 4 + c]; }
        for (int q = 0; q < 9; ++q) { int r = n + q - 4; if (r >= 0 && r < NPTS) cv += wx9[q] * Gg[r * 4 + c]; }
        float raw = GgC[n * 4 + c] + S1[c] * (1.f + bsum) - Dg[n * 4 + c] / md + cv;
        h4[n * 4 + c] = (raw - mn * 5.f * S1[c]) * inv;
    }
}

// ---------- h0 = leaky(h4 @ Woff + boff) -> f16 ----------
__global__ void k_woff(const float* __restrict__ h4, const float* __restrict__ Woff,
                       const float* __restrict__ boff, const float* __restrict__ pw,
                       half_t* __restrict__ h0) {
    int id = blockIdx.x * blockDim.x + threadIdx.x;
    if (id >= NPTS * DIMF) return;
    int n = id / DIMF, f = id % DIMF;
    float s = boff[f];
    for (int c = 0; c < 4; ++c) s += h4[n * 4 + c] * Woff[c * DIMF + f];
    s = s > 0.f ? s : pw[0] * s;
    h0[id] = (half_t)s;
}

// ---------- out = X2 @ Wf3 + bf3 ----------
__global__ void k_wf3(const half_t* __restrict__ X2, const float* __restrict__ Wf3,
                      const float* __restrict__ bf3, float* __restrict__ out) {
    int id = blockIdx.x * blockDim.x + threadIdx.x;
    if (id >= NPTS * 4) return;
    int n = id / 4, c = id % 4;
    float s = bf3[c];
    for (int k = 0; k < HIDF; ++k) s += (float)X2[n * HIDF + k] * Wf3[k * 4 + c];
    out[id] = s;
}

static inline int cdiv(long long a, long long b) { return (int)((a + b - 1) / b); }

extern "C" void kernel_launch(void* const* d_in, const int* in_sizes, int n_in,
                              void* d_out, int out_size, void* d_ws, size_t ws_size,
                              hipStream_t stream) {
    const float* x1   = (const float*)d_in[0];
    const float* x2   = (const float*)d_in[1];
    const float* Wk   = (const float*)d_in[2];
    const float* bk   = (const float*)d_in[3];
    const float* Wg1  = (const float*)d_in[4];
    const float* bg1  = (const float*)d_in[5];
    const float* Wg2  = (const float*)d_in[6];
    const float* bg2  = (const float*)d_in[7];
    const float* Wq   = (const float*)d_in[8];
    const float* bq   = (const float*)d_in[9];
    const float* Wkey = (const float*)d_in[10];
    const float* bkey = (const float*)d_in[11];
    const float* Woff = (const float*)d_in[12];
    const float* boff = (const float*)d_in[13];
    const float* pw   = (const float*)d_in[14];
    const float* Wf1  = (const float*)d_in[15];
    const float* bf1  = (const float*)d_in[16];
    const float* Wf2  = (const float*)d_in[17];
    const float* bf2  = (const float*)d_in[18];
    const float* Wf3  = (const float*)d_in[19];
    const float* bf3  = (const float*)d_in[20];
    const float* Wcy3 = (const float*)d_in[21];
    const float* bcy3 = (const float*)d_in[22];
    const float* Wcx3 = (const float*)d_in[23];
    const float* bcx3 = (const float*)d_in[24];
    const float* Wcy5 = (const float*)d_in[25];
    const float* bcy5 = (const float*)d_in[26];
    const float* Wcx5 = (const float*)d_in[27];
    const float* bcx5 = (const float*)d_in[28];
    const float* Wcy9 = (const float*)d_in[29];
    const float* bcy9 = (const float*)d_in[30];
    const float* Wcx9 = (const float*)d_in[31];
    const float* bcx9 = (const float*)d_in[32];
    float* out = (float*)d_out;

    // bump allocator over d_ws (32B aligned so paired v8h loads are natural)
    size_t cur = 0;
    float* base = (float*)d_ws;
    auto af = [&](size_t n) -> float* {
        float* p = base + cur;
        cur += (n + 7) & ~(size_t)7;
        return p;
    };
    auto ah = [&](size_t n) -> half_t* { return (half_t*)af((n + 1) / 2 + 8); };

    const size_t Nl = NPTS;
    float* dx8  = af(Nl * 8);
    float* maxd = af(Nl);
    float* OtC  = af(Nl * 4);
    float* S1   = af(4);
    float* Gg   = af(Nl * 4);
    float* GgC  = af(Nl * 4);
    float* Dg   = af(Nl * 4);
    float* h4   = af(Nl * 4);
    unsigned* mm = (unsigned*)af(4);
    float* degf = af(Nl);
    float* dinv = af(Nl);
    float* xw   = af(Nl * 128);
    float* agg  = af(Nl * 128);
    float* h1   = af(Nl * 64);
    float* emb  = af(Nl * 384);
    int*   knn  = (int*)af(Nl * 9);
    half_t* embH = ah(Nl * 384);
    half_t* WqT  = ah(128 * 384);   // [N=128][K=384]
    half_t* WkT  = ah(128 * 384);
    half_t* Wf1T = ah(256 * 128);   // [N=256][K=128]
    half_t* Wf2T = ah(256 * 256);   // [N=256][K=256]
    half_t* Qh   = ah(Nl * 128);
    half_t* Kh   = ah(Nl * 128);
    half_t* h0H  = ah(Nl * 128);
    half_t* X1H  = ah(Nl * 256);
    half_t* X2H  = ah(Nl * 256);
    float* dist = af(Nl * Nl);
    float* Ab   = af(Nl * Nl);

    const int B = 256;

    k_build<<<cdiv(NPTS, B), B, 0, stream>>>(x1, x2, Wk, bk, dx8);
    k_dist<<<NPTS, B, 0, stream>>>(dx8, dist, maxd);
    k_knn<<<NPTS, B, 0, stream>>>(dist, knn);

    // three GCN towers (k = 3,5,9), results into emb columns
    const int kks[3] = {3, 5, 9};
    for (int gi = 0; gi < 3; ++gi) {
        int kk = kks[gi], off = gi * 128;
        k_fill<<<cdiv(NPTS, B), B, 0, stream>>>(degf, 1.0f, NPTS);
        k_deg<<<cdiv((long long)NPTS * kk, B), B, 0, stream>>>(knn, kk, degf);
        k_dinv<<<cdiv(NPTS, B), B, 0, stream>>>(degf, dinv);
        k_gemm_f32<<<cdiv((long long)NPTS * 64, B), B, 0, stream>>>(dx8, Wg1, xw, 8, 64);
        k_fill<<<cdiv((long long)NPTS * 64, B), B, 0, stream>>>(agg, 0.f, NPTS * 64);
        k_agg<<<cdiv((long long)NPTS * (kk + 1) * 64, B), B, 0, stream>>>(xw, knn, kk, dinv, agg, 64);
        k_bias_relu<<<cdiv((long long)NPTS * 64, B), B, 0, stream>>>(agg, bg1, h1, 64);
        k_gemm_f32<<<cdiv((long long)NPTS * 128, B), B, 0, stream>>>(h1, Wg2, xw, 64, 128);
        k_fill<<<cdiv((long long)NPTS * 128, B), B, 0, stream>>>(agg, 0.f, NPTS * 128);
        k_agg<<<cdiv((long long)NPTS * (kk + 1) * 128, B), B, 0, stream>>>(xw, knn, kk, dinv, agg, 128);
        k_lsm<<<NPTS, 128, 0, stream>>>(agg, bg2, emb, off);
    }

    // f16 conversions (activations straight, weights transposed to [N][K])
    k_cvt_h<<<cdiv((long long)NPTS * 384, B), B, 0, stream>>>(emb, embH, NPTS * 384);
    k_cvt_t<<<cdiv(384 * 128, B), B, 0, stream>>>(Wq,   WqT,  384, 128);
    k_cvt_t<<<cdiv(384 * 128, B), B, 0, stream>>>(Wkey, WkT,  384, 128);
    k_cvt_t<<<cdiv(128 * 256, B), B, 0, stream>>>(Wf1,  Wf1T, 128, 256);
    k_cvt_t<<<cdiv(256 * 256, B), B, 0, stream>>>(Wf2,  Wf2T, 256, 256);

    auto wmma_blocks = [&](int M, int Nn, int NT) {
        long long waves = (long long)(M / 16) * (Nn / 16 / NT);
        return cdiv(waves * 32, B);
    };

    // Q, K projections (WMMA f16, NT=2)
    k_wmma<2, 0, 1, 1><<<wmma_blocks(NPTS, 128, 2), B, 0, stream>>>(embH, WqT, bq, 1.f,
                                                                    nullptr, Qh, NPTS, 128, 384);
    k_wmma<2, 0, 1, 1><<<wmma_blocks(NPTS, 128, 2), B, 0, stream>>>(embH, WkT, bkey, 1.f,
                                                                    nullptr, Kh, NPTS, 128, 384);
    // scores = (Q @ K^T) / sqrt(128)  (WMMA, NT=4)
    k_wmma<4, 0, 0, 0><<<wmma_blocks(NPTS, NPTS, 4), B, 0, stream>>>(
        Qh, Kh, nullptr, 0.08838834764832f, Ab, nullptr, NPTS, NPTS, 128);
    k_rowdiv<<<NPTS, B, 0, stream>>>(Ab);

    // min/max of A_cat + fused matvec pieces
    k_mm_init<<<1, 1, 0, stream>>>(mm);
    k_minmax<<<NPTS, B, 0, stream>>>(Ab, dist, maxd, Wcy3, bcy3, Wcx3, bcx3, Wcy5, bcy5, Wcx5,
                                     bcx5, Wcy9, bcy9, Wcx9, bcx9, mm);
    k_s1<<<1, B, 0, stream>>>(dx8, S1);
    k_otc<<<cdiv(NPTS, B), B, 0, stream>>>(dx8, Wcy3, Wcy5, Wcy9, OtC);
    k_rowmm<<<NPTS, B, 0, stream>>>(Ab, dist, dx8, OtC, Gg, GgC, Dg);
    k_raw<<<cdiv(NPTS, B), B, 0, stream>>>(Gg, GgC, Dg, maxd, S1, mm, Wcx3, Wcx5, Wcx9, bcy3,
                                           bcx3, bcy5, bcx5, bcy9, bcx9, h4);

    // MLP tail (WMMA for the two big layers, NT=2)
    k_woff<<<cdiv((long long)NPTS * DIMF, B), B, 0, stream>>>(h4, Woff, boff, pw, h0H);
    k_wmma<2, 1, 1, 1><<<wmma_blocks(NPTS, HIDF, 2), B, 0, stream>>>(h0H, Wf1T, bf1, 1.f,
                                                                     nullptr, X1H, NPTS, HIDF, DIMF);
    k_wmma<2, 1, 1, 1><<<wmma_blocks(NPTS, HIDF, 2), B, 0, stream>>>(X1H, Wf2T, bf2, 1.f,
                                                                     nullptr, X2H, NPTS, HIDF, HIDF);
    k_wf3<<<cdiv((long long)NPTS * 4, B), B, 0, stream>>>(X2H, Wf3, bf3, out);

    (void)in_sizes; (void)n_in; (void)out_size; (void)ws_size;
}
